// MultiHeadContextualBiasedAttention_46583215293069
// MI455X (gfx1250) — matrix operand order repaired and checked by reading, not compiled, not run on hardware
//
#include <hip/hip_runtime.h>
#include <hip/hip_bf16.h>

typedef __attribute__((ext_vector_type(16))) __bf16 v16bf;
typedef __attribute__((ext_vector_type(8)))  __bf16 v8bf;
typedef __attribute__((ext_vector_type(8)))  float  v8f;
typedef __attribute__((ext_vector_type(4)))  int    v4i;

#define D_MODEL 1024
#define D_HEAD  64
#define N_HEAD  16
#define BB      2
#define T1_     1024
#define T2_     2048

// ---- helpers -------------------------------------------------------------

__device__ __forceinline__ unsigned short f2bf(float f) {
    unsigned u = __float_as_uint(f);
    unsigned r = ((u >> 16) & 1u) + 0x7FFFu;   // round-to-nearest-even
    return (unsigned short)((u + r) >> 16);
}

__device__ __forceinline__ v8bf ld8(const unsigned short* p) {
    return *reinterpret_cast<const v8bf*>(p);
}

__device__ __forceinline__ v16bf cat16(v8bf lo, v8bf hi) {
    v16bf r;
#pragma unroll
    for (int i = 0; i < 8; ++i) { r[i] = lo[i]; r[i + 8] = hi[i]; }
    return r;
}

// 16-byte async global->LDS copy (CDNA5 ASYNCcnt path), with safe fallback
__device__ __forceinline__ void async_cp16(const unsigned short* g, unsigned short* l) {
#if __has_builtin(__builtin_amdgcn_global_load_async_to_lds_b128)
    __builtin_amdgcn_global_load_async_to_lds_b128(
        (__attribute__((address_space(1))) v4i*)g,
        (__attribute__((address_space(3))) v4i*)l, 0, 0);
#else
    *reinterpret_cast<uint4*>(l) = *reinterpret_cast<const uint4*>(g);
#endif
}

__device__ __forceinline__ void wait_async_prev() {   // previous chunk landed
#if __has_builtin(__builtin_amdgcn_s_wait_asynccnt)
    __builtin_amdgcn_s_wait_asynccnt(16);
#else
    asm volatile("s_wait_asynccnt 0x10" ::: "memory");
#endif
}

__device__ __forceinline__ void wait_async_all() {
#if __has_builtin(__builtin_amdgcn_s_wait_asynccnt)
    __builtin_amdgcn_s_wait_asynccnt(0);
#else
    asm volatile("s_wait_asynccnt 0x0" ::: "memory");
#endif
}

// ---- f32 -> bf16 conversion kernels -------------------------------------

__global__ void __launch_bounds__(256)
cvt_bf16(const float* __restrict__ s, unsigned short* __restrict__ d, int n) {
    int i = blockIdx.x * 256 + threadIdx.x;
    if (i < n) d[i] = f2bf(s[i]);
}

// W is [K=1024][N=1024]; write WT[n][k] bf16 so GEMM B-operands are K-contiguous
__global__ void __launch_bounds__(256)
cvt_t_bf16(const float* __restrict__ s, unsigned short* __restrict__ d) {
    int i = blockIdx.x * 256 + threadIdx.x;    // i = n*1024 + k
    int n = i >> 10, k = i & 1023;
    d[i] = f2bf(s[(size_t)k * D_MODEL + n]);
}

// ---- generic bf16 WMMA GEMM: Y[M][1024] = X[M][1024] @ WT^T ---------------
// mode 0: write bf16 head-major [b,h,t,64]      (Q / K projections)
// mode 2: write bf16 transposed [b,h,64,T]      (V projection, V^T)
// mode 3: write f32 [M][1024] + bias1d          (output projection)

__global__ void __launch_bounds__(256)
gemm_bf16_wmma(const unsigned short* __restrict__ X,
               const unsigned short* __restrict__ WT,
               unsigned short* __restrict__ Ybf,
               float* __restrict__ Yf,
               const float* __restrict__ bias1d,
               int M, int T, int mode)
{
    const int wave = (blockIdx.x * blockDim.x + threadIdx.x) >> 5;
    const int lane = threadIdx.x & 31;
    const int row  = lane & 15;
    const int half = lane >> 4;
    const int kg   = half * 8;

    const int ncg = D_MODEL / 64;              // 16 groups of 64 columns
    const int m0  = (wave / ncg) * 16;
    const int n0  = (wave % ncg) * 64;
    if (m0 >= M) return;

    v8f acc[4];
#pragma unroll
    for (int t = 0; t < 4; ++t)
#pragma unroll
        for (int i = 0; i < 8; ++i) acc[t][i] = 0.f;

    const unsigned short* xrow  = X  + (size_t)(m0 + row) * D_MODEL;
    const unsigned short* wbase = WT + (size_t)(n0 + row) * D_MODEL + half * 16;

    // Fully unrolled k-loop: one straight-line block so every step uses fresh
    // registers (no cross-iteration rotation movs) and the scheduler can hoist
    // loads across earlier WMMAs.
#pragma unroll
    for (int k0 = 0; k0 < D_MODEL; k0 += 32) {
        v16bf a = cat16(ld8(xrow + k0 + 2 * kg), ld8(xrow + k0 + 2 * kg + 8));
#pragma unroll
        for (int t = 0; t < 4; ++t) {
            const unsigned short* wr = wbase + (size_t)t * 16 * D_MODEL + k0;
            v16bf bm = cat16(ld8(wr), ld8(wr + 8));
            acc[t] = __builtin_amdgcn_wmma_f32_16x16x32_bf16(
                false, a, false, bm, (short)0, acc[t], false, false);
        }
    }

#pragma unroll
    for (int t = 0; t < 4; ++t) {
#pragma unroll
        for (int r = 0; r < 8; ++r) {
            float v  = acc[t][r];
            int   n  = n0 + 16 * t + row;
            int   mg = m0 + r + half * 8;           // C/D layout: M = r + half*8
            if (mode == 3) {
                Yf[(size_t)mg * D_MODEL + n] = v + bias1d[n];
            } else {
                int b = mg / T, tt = mg % T;
                int h = n >> 6, d = n & 63;
                if (mode == 0)
                    Ybf[(((size_t)(b * N_HEAD + h) * T) + tt) * D_HEAD + d] = f2bf(v);
                else
                    Ybf[(((size_t)(b * N_HEAD + h) * D_HEAD) + d) * (size_t)T + tt] = f2bf(v);
            }
        }
    }
}

// ---- fused flash attention: one wave per 16 query rows --------------------
// K/V chunks are double-buffered in LDS via async global->LDS copies.

#define AW 4   // waves per block

__global__ void __launch_bounds__(32 * AW)
attn_fused_wmma(const unsigned short* __restrict__ Q,    // [B,H,T1,64] bf16
                const unsigned short* __restrict__ K,    // [B,H,T2,64] bf16
                const unsigned short* __restrict__ Vt,   // [B,H,64,T2] bf16
                const float* __restrict__ bias,          // [B,H,T1,T2] f32
                const int* __restrict__ mask,            // [B,1,T1,T2]
                unsigned short* __restrict__ O)          // [B,T1,1024] bf16
{
    __shared__ __align__(16) unsigned short ldsK[AW][2][32 * 64]; // [key][d]
    __shared__ __align__(16) unsigned short ldsV[AW][2][64 * 32]; // [d][key]
    __shared__ __align__(16) unsigned short ldsP[AW][16][48];

    const int w    = threadIdx.x >> 5;
    const int wave = blockIdx.x * AW + w;
    const int lane = threadIdx.x & 31;
    const int row  = lane & 15;
    const int half = lane >> 4;
    const int kg   = half * 8;

    const int qtph = T1_ / 16;                 // 64 q-tiles per (b,h)
    const int bh = wave / qtph;
    const int q0 = (wave % qtph) * 16;
    const int b  = bh >> 4, h = bh & 15;

    // Q A-operands, register-resident (k-steps 0..31, 32..63 of d)
    const unsigned short* qbase = Q + ((size_t)bh * T1_ + q0) * D_HEAD;
    v16bf aq[2];
#pragma unroll
    for (int c = 0; c < 2; ++c) {
        const unsigned short* qr = qbase + (size_t)row * D_HEAD + 32 * c + kg;
        aq[c] = cat16(ld8(qr + kg), ld8(qr + kg + 8));
    }

    v8f o[4];
#pragma unroll
    for (int t = 0; t < 4; ++t)
#pragma unroll
        for (int i = 0; i < 8; ++i) o[t][i] = 0.f;
    float mrow[8], lrow[8];
#pragma unroll
    for (int r = 0; r < 8; ++r) { mrow[r] = -1e30f; lrow[r] = 0.f; }

    const float scale = 0.125f;                              // 1/sqrt(64)
    const float* bias_base = bias + ((size_t)bh * T1_ + q0) * T2_;
    const int*   mask_base = mask + ((size_t)b  * T1_ + q0) * T2_;
    const unsigned short* Kg  = K  + (size_t)bh * T2_ * D_HEAD;
    const unsigned short* Vtg = Vt + (size_t)bh * D_HEAD * T2_;

    // issue async copy of a 32-key chunk into LDS buffer `buf` (16 x b128/lane)
    auto issue_chunk = [&](int j, int buf) {
        // K rows j..j+31: 4KB linear
#pragma unroll
        for (int i = 0; i < 8; ++i) {
            int e = i * 256 + lane * 8;                       // element offset
            async_cp16(Kg + (size_t)j * D_HEAD + e, &ldsK[w][buf][e]);
        }
        // V^T rows: 64 rows x 64B (strided by T2)
#pragma unroll
        for (int i = 0; i < 8; ++i) {
            int tIdx = i * 32 + lane;                         // 0..255
            int dr   = tIdx >> 2;                             // d row 0..63
            int seg  = tIdx & 3;                              // 16B segment
            async_cp16(Vtg + (size_t)dr * T2_ + j + seg * 8,
                       &ldsV[w][buf][dr * 32 + seg * 8]);
        }
    };

    issue_chunk(0, 0);

#pragma unroll 2   // makes `buf` a compile-time constant per unrolled body
    for (int j = 0; j < T2_; j += 32) {
        const int buf = (j >> 5) & 1;
        if (j + 32 < T2_) {
            issue_chunk(j + 32, buf ^ 1);
            wait_async_prev();        // current buffer's 16 transfers complete
        } else {
            wait_async_all();
        }

        // bias + mask for this chunk (independent of WMMAs; loads early)
        float bv[8][2];
        int   mk[8][2];
#pragma unroll
        for (int r = 0; r < 8; ++r) {
            size_t off = (size_t)(r + half * 8) * T2_ + j + row;
#pragma unroll
            for (int g = 0; g < 2; ++g) {
                bv[r][g] = bias_base[off + 16 * g];
                mk[r][g] = mask_base[off + 16 * g];
            }
        }

        // ---- S = Q K^T from LDS (all operands fetched, then 4 WMMAs) ----
        v16bf bk[2][2];
#pragma unroll
        for (int g = 0; g < 2; ++g)
#pragma unroll
            for (int c = 0; c < 2; ++c) {
                const unsigned short* kr =
                    &ldsK[w][buf][(16 * g + row) * D_HEAD + 32 * c + half * 16];
                bk[g][c] = cat16(ld8(kr), ld8(kr + 8));
            }
        v8f s[2];
#pragma unroll
        for (int g = 0; g < 2; ++g) {
#pragma unroll
            for (int i = 0; i < 8; ++i) s[g][i] = 0.f;
#pragma unroll
            for (int c = 0; c < 2; ++c)
                s[g] = __builtin_amdgcn_wmma_f32_16x16x32_bf16(
                    false, aq[c], false, bk[g][c], (short)0, s[g], false, false);
        }

        // ---- bias, scale, mask ----
#pragma unroll
        for (int r = 0; r < 8; ++r)
#pragma unroll
            for (int g = 0; g < 2; ++g) {
                float sv = (s[g][r] + bv[r][g]) * scale;
                if (mk[r][g]) sv = -1e9f;
                s[g][r] = sv;
            }

        // ---- online softmax (row reductions across 16-lane halves) ----
#pragma unroll
        for (int r = 0; r < 8; ++r) {
            float rm = fmaxf(s[0][r], s[1][r]);
#pragma unroll
            for (int d = 1; d < 16; d <<= 1)
                rm = fmaxf(rm, __shfl_xor(rm, d, 16));
            float mn = fmaxf(mrow[r], rm);
            float al = __expf(mrow[r] - mn);
            float p0 = (s[0][r] < -5e8f) ? 0.f : __expf(s[0][r] - mn);
            float p1 = (s[1][r] < -5e8f) ? 0.f : __expf(s[1][r] - mn);
            float rs = p0 + p1;
#pragma unroll
            for (int d = 1; d < 16; d <<= 1)
                rs += __shfl_xor(rs, d, 16);
            lrow[r] = lrow[r] * al + rs;
            mrow[r] = mn;
#pragma unroll
            for (int t = 0; t < 4; ++t) o[t][r] *= al;
            ldsP[w][r + half * 8][row]      = f2bf(p0);
            ldsP[w][r + half * 8][row + 16] = f2bf(p1);
        }

        asm volatile("s_wait_dscnt 0x0" ::: "memory");

        // ---- P (A layout) @ V (B layout, from LDS) ----
        const unsigned short* pr = &ldsP[w][row][0];
        v16bf pA = cat16(ld8(pr + 2 * kg), ld8(pr + 2 * kg + 8));
        v16bf bvv[4];
#pragma unroll
        for (int t = 0; t < 4; ++t) {
            const unsigned short* vr =
                &ldsV[w][buf][(16 * t + row) * 32 + half * 16];
            bvv[t] = cat16(ld8(vr), ld8(vr + 8));
        }
#pragma unroll
        for (int t = 0; t < 4; ++t)
            o[t] = __builtin_amdgcn_wmma_f32_16x16x32_bf16(
                false, pA, false, bvv[t], (short)0, o[t], false, false);
    }

    // ---- normalize and write O[b, q, h*64 + d] as bf16 ----
#pragma unroll
    for (int r = 0; r < 8; ++r) {
        float inv = 1.f / fmaxf(lrow[r], 1e-30f);
        int q = q0 + r + half * 8;
        unsigned short* orow = O + ((size_t)b * T1_ + q) * D_MODEL + h * D_HEAD;
#pragma unroll
        for (int t = 0; t < 4; ++t)
            orow[16 * t + row] = f2bf(o[t][r] * inv);
    }
}

// ---- host-side launch -----------------------------------------------------

extern "C" void kernel_launch(void* const* d_in, const int* in_sizes, int n_in,
                              void* d_out, int out_size, void* d_ws, size_t ws_size,
                              hipStream_t stream) {
    const float* x    = (const float*)d_in[0];
    const float* ctx  = (const float*)d_in[1];
    const float* bias = (const float*)d_in[2];
    const int*   mask = (const int*)  d_in[3];
    const float* Wq   = (const float*)d_in[4];
    const float* Wk   = (const float*)d_in[5];
    const float* Wv   = (const float*)d_in[6];
    const float* Wo   = (const float*)d_in[7];
    const float* bo   = (const float*)d_in[8];
    float* out = (float*)d_out;

    char* ws = (char*)d_ws;
    size_t off = 0;
    auto carve = [&](size_t bytes) -> unsigned short* {
        char* p = ws + off;
        off += (bytes + 255) & ~(size_t)255;
        return (unsigned short*)p;
    };
    const size_t E = sizeof(unsigned short);
    unsigned short* xb  = carve((size_t)BB * T1_ * D_MODEL * E);
    unsigned short* cb  = carve((size_t)BB * T2_ * D_MODEL * E);
    unsigned short* wqT = carve((size_t)D_MODEL * D_MODEL * E);
    unsigned short* wkT = carve((size_t)D_MODEL * D_MODEL * E);
    unsigned short* wvT = carve((size_t)D_MODEL * D_MODEL * E);
    unsigned short* woT = carve((size_t)D_MODEL * D_MODEL * E);
    unsigned short* Qb  = carve((size_t)BB * N_HEAD * T1_ * D_HEAD * E);
    unsigned short* Kb  = carve((size_t)BB * N_HEAD * T2_ * D_HEAD * E);
    unsigned short* Vtb = carve((size_t)BB * N_HEAD * D_HEAD * T2_ * E);
    unsigned short* Ob  = carve((size_t)BB * T1_ * D_MODEL * E);

    const int nx = BB * T1_ * D_MODEL;   // 2,097,152
    const int nc = BB * T2_ * D_MODEL;   // 4,194,304
    const int nw = D_MODEL * D_MODEL;    // 1,048,576

    cvt_bf16  <<<nx / 256, 256, 0, stream>>>(x,   xb, nx);
    cvt_bf16  <<<nc / 256, 256, 0, stream>>>(ctx, cb, nc);
    cvt_t_bf16<<<nw / 256, 256, 0, stream>>>(Wq, wqT);
    cvt_t_bf16<<<nw / 256, 256, 0, stream>>>(Wk, wkT);
    cvt_t_bf16<<<nw / 256, 256, 0, stream>>>(Wv, wvT);
    cvt_t_bf16<<<nw / 256, 256, 0, stream>>>(Wo, woT);

    // projections: wave tiles = (M/16)*16, 8 waves per 256-thread block
    gemm_bf16_wmma<<<256, 256, 0, stream>>>(xb, wqT, Qb,  nullptr, nullptr,
                                            BB * T1_, T1_, 0);
    gemm_bf16_wmma<<<512, 256, 0, stream>>>(cb, wkT, Kb,  nullptr, nullptr,
                                            BB * T2_, T2_, 0);
    gemm_bf16_wmma<<<512, 256, 0, stream>>>(cb, wvT, Vtb, nullptr, nullptr,
                                            BB * T2_, T2_, 2);

    // fused attention: 2048 wave-tiles (32 bh * 64 q-tiles), 4 waves/block
    attn_fused_wmma<<<512, 32 * AW, 0, stream>>>(Qb, Kb, Vtb, bias, mask, Ob);

    // output projection + b_out, f32 result
    gemm_bf16_wmma<<<256, 256, 0, stream>>>(Ob, woT, nullptr, out, bo,
                                            BB * T1_, T1_, 3);
}
